// BinaryNet_79912161509544
// MI455X (gfx1250) — 8-bit matrix operand order repaired and checked by reading, compiled, not run on hardware
//
#include <hip/hip_runtime.h>
#include <stdint.h>


typedef int v8i __attribute__((ext_vector_type(8)));
typedef int v4i __attribute__((ext_vector_type(4)));

#define BN_EPS 1e-3f
#define LDS_STRIDE 144
#define BUFSZ (64 * LDS_STRIDE)

__global__ __launch_bounds__(256) void bgemm_wmma(
    const int8_t* __restrict__ in, const int8_t* __restrict__ w,
    void* __restrict__ out, int M, int NP,
    int H, int W, int Cin, int ksize, int out32)
{
  __shared__ __align__(16) int8_t As[2 * BUFSZ];
  __shared__ __align__(16) int8_t Bs[2 * BUFSZ];

  const int t    = threadIdx.x;
  const int lane = t & 31;
  const int wave = t >> 5;
  const int mw   = wave & 3;
  const int nw   = wave >> 2;
  const int mBase = blockIdx.x * 64;
  const int nBase = blockIdx.y * 64;

  const int lr  = t >> 2;
  const int seg = t & 3;

  const int am = mBase + lr;
  const bool am_ok = am < M;
  int ab = 0, ay = 0, ax = 0;
  if (am_ok) {
    const int hw = H * W;
    ab = am / hw;
    const int r2 = am - ab * hw;
    ay = r2 / W;
    ax = r2 - ay * W;
  }
  const int pad = ksize >> 1;
  const int K   = ksize * ksize * Cin;

  const int8_t* browg = w + (size_t)(nBase + lr) * (size_t)K + seg * 16;

  const v4i vzero = {0, 0, 0, 0};
  v8i acc0 = {0, 0, 0, 0, 0, 0, 0, 0};
  v8i acc1 = {0, 0, 0, 0, 0, 0, 0, 0};

  const int m16  = lane & 15;
  const int half = lane >> 4;
  const int8_t* const arowF  = As + (mw * 16 + m16) * LDS_STRIDE + half * 16;
  const int8_t* const brow0F = Bs + (nw * 32 + m16) * LDS_STRIDE + half * 16;
  const int8_t* const brow1F = brow0F + 16 * LDS_STRIDE;
  int8_t* const astF = As + lr * LDS_STRIDE + seg * 16;
  int8_t* const bstF = Bs + lr * LDS_STRIDE + seg * 16;

  const int stagesPerTap = Cin >> 7;
  const int nstages = ksize * ksize * stagesPerTap;

  int ky = 0, kx = 0, c0 = 0;
  bool vrow = false;
  const int8_t* arowg = in;
  const int8_t* bsegg = browg;

  auto setup_tap = [&]() {
    const int iy = ay + ky - pad;
    const int ix = ax + kx - pad;
    vrow = am_ok && iy >= 0 && iy < H && ix >= 0 && ix < W;
    arowg = in + (size_t)((ab * H + iy) * W + ix) * (size_t)Cin + seg * 16;
    bsegg = browg + (ky * ksize + kx) * Cin;
  };
  v4i ra0, ra1, rb0, rb1;
  auto load_stage = [&]() {
    ra0 = vzero; ra1 = vzero;
    if (vrow) {
      ra0 = *(const v4i*)(arowg + c0);
      ra1 = *(const v4i*)(arowg + c0 + 64);
    }
    rb0 = *(const v4i*)(bsegg + c0);
    rb1 = *(const v4i*)(bsegg + c0 + 64);
  };
  auto advance = [&]() {
    c0 += 128;
    if (c0 == Cin) {
      c0 = 0;
      if (++kx == ksize) { kx = 0; ++ky; }
      if (ky < ksize) setup_tap();
    }
  };

  setup_tap();
  load_stage();
  advance();
  *(v4i*)(astF)      = ra0;
  *(v4i*)(astF + 64) = ra1;
  *(v4i*)(bstF)      = rb0;
  *(v4i*)(bstF + 64) = rb1;
  if (nstages > 1) { load_stage(); advance(); }
  __syncthreads();

  for (int s = 0; s < nstages; ++s) {
    const int cur = s & 1;
    const int nxt = cur ^ 1;

    if (s + 1 < nstages) {
      *(v4i*)(astF + nxt * BUFSZ)      = ra0;
      *(v4i*)(astF + nxt * BUFSZ + 64) = ra1;
      *(v4i*)(bstF + nxt * BUFSZ)      = rb0;
      *(v4i*)(bstF + nxt * BUFSZ + 64) = rb1;
    }
    if (s + 2 < nstages) { load_stage(); advance(); }

    const int co = cur * BUFSZ;
    #pragma unroll
    for (int kk = 0; kk < 128; kk += 64) {
      union { v8i v; unsigned long long d[4]; } Af;
      {
        const int8_t* ap = arowF + co + kk;
        Af.d[0] = *(const unsigned long long*)(ap +  0);
        Af.d[1] = *(const unsigned long long*)(ap +  8);
        Af.d[2] = *(const unsigned long long*)(ap + 32);
        Af.d[3] = *(const unsigned long long*)(ap + 40);
      }
      union { v8i v; v4i q[2]; } Bf0, Bf1;
      {
        const int8_t* bp0 = brow0F + co + kk;
        Bf0.q[0] = *(const v4i*)(bp0 +  0);
        Bf0.q[1] = *(const v4i*)(bp0 + 32);
        const int8_t* bp1 = brow1F + co + kk;
        Bf1.q[0] = *(const v4i*)(bp1 +  0);
        Bf1.q[1] = *(const v4i*)(bp1 + 32);
      }
      acc0 = __builtin_amdgcn_wmma_i32_16x16x64_iu8(true, Af.v, true, Bf0.v,
                                                    acc0, false, false);
      acc1 = __builtin_amdgcn_wmma_i32_16x16x64_iu8(true, Af.v, true, Bf1.v,
                                                    acc1, false, false);
    }

    __syncthreads();
  }

  #pragma unroll
  for (int f = 0; f < 2; ++f) {
    const v8i a = f ? acc1 : acc0;
    const int col = nBase + nw * 32 + f * 16 + m16;
    #pragma unroll
    for (int v = 0; v < 8; ++v) {
      const int m = mBase + mw * 16 + half * 8 + v;
      if (m < M) {
        const size_t o = (size_t)m * (size_t)NP + col;
        if (out32) ((int32_t*)out)[o] = a[v];
        else       ((int16_t*)out)[o] = (int16_t)a[v];
      }
    }
  }
}

__global__ void conv1_bn_sign(const float* __restrict__ x,
                              const float* __restrict__ w,
                              const float* __restrict__ beta,
                              const float* __restrict__ mean,
                              const float* __restrict__ var,
                              int8_t* __restrict__ out, int total)
{
  const int tid = blockIdx.x * blockDim.x + threadIdx.x;
  if (tid >= total) return;
  const int co = tid & 127;
  int r = tid >> 7;
  const int px = r % 96; r /= 96;
  const int py = r % 96;
  const int b  = r / 96;
  float s = 0.f;
  for (int ky = 0; ky < 3; ++ky) {
    const int iy = py + ky - 1;
    if (iy < 0 || iy >= 96) continue;
    for (int kx = 0; kx < 3; ++kx) {
      const int ix = px + kx - 1;
      if (ix < 0 || ix >= 96) continue;
      const float* xp = x + (size_t)((b * 96 + iy) * 96 + ix) * 3;
      const float* wp = w + (size_t)((ky * 3 + kx) * 3) * 128 + co;
      #pragma unroll
      for (int ci = 0; ci < 3; ++ci)
        s += xp[ci] * (wp[ci * 128] >= 0.f ? 1.f : -1.f);
    }
  }
  const float y = (s - mean[co]) * rsqrtf(var[co] + BN_EPS) + beta[co];
  out[tid] = (y >= 0.f) ? (int8_t)1 : (int8_t)-1;
}

__global__ void transpose_sign(const float* __restrict__ w,
                               int8_t* __restrict__ out,
                               int K, int N, long total)
{
  const long tid = (long)blockIdx.x * blockDim.x + threadIdx.x;
  if (tid >= total) return;
  const int k = (int)(tid % K);
  const int n = (int)(tid / K);
  out[tid] = (n < N) ? ((w[(size_t)k * N + n] >= 0.f) ? (int8_t)1 : (int8_t)-1)
                     : (int8_t)0;
}

__global__ void pool_bn_sign(const int16_t* __restrict__ acc,
                             int8_t* __restrict__ out,
                             const float* __restrict__ beta,
                             const float* __restrict__ mean,
                             const float* __restrict__ var,
                             int Ho, int Wo, int C, int total)
{
  const int tid = blockIdx.x * blockDim.x + threadIdx.x;
  if (tid >= total) return;
  const int c = tid % C;
  int r = tid / C;
  const int x = r % Wo; r /= Wo;
  const int y = r % Ho;
  const int b = r / Ho;
  const int Wi = Wo * 2;
  const int16_t* p =
      acc + ((size_t)((b * (Ho * 2) + y * 2) * Wi + x * 2)) * (size_t)C + c;
  int v = p[0];
  int v1 = p[C];                     if (v1 > v) v = v1;
  v1 = p[(size_t)Wi * C];            if (v1 > v) v = v1;
  v1 = p[(size_t)Wi * C + C];        if (v1 > v) v = v1;
  const float yv = ((float)v - mean[c]) * rsqrtf(var[c] + BN_EPS) + beta[c];
  out[tid] = (yv >= 0.f) ? (int8_t)1 : (int8_t)-1;
}

__global__ void bn_sign16(const int16_t* __restrict__ acc,
                          int8_t* __restrict__ out,
                          const float* __restrict__ beta,
                          const float* __restrict__ mean,
                          const float* __restrict__ var, int C, int total)
{
  const int tid = blockIdx.x * blockDim.x + threadIdx.x;
  if (tid >= total) return;
  const int c = tid % C;
  const float yv = ((float)acc[tid] - mean[c]) * rsqrtf(var[c] + BN_EPS) + beta[c];
  out[tid] = (yv >= 0.f) ? (int8_t)1 : (int8_t)-1;
}

__global__ void bn_sign32(const int32_t* __restrict__ acc,
                          int8_t* __restrict__ out,
                          const float* __restrict__ beta,
                          const float* __restrict__ mean,
                          const float* __restrict__ var, int C, int total)
{
  const int tid = blockIdx.x * blockDim.x + threadIdx.x;
  if (tid >= total) return;
  const int c = tid % C;
  const float yv = ((float)acc[tid] - mean[c]) * rsqrtf(var[c] + BN_EPS) + beta[c];
  out[tid] = (yv >= 0.f) ? (int8_t)1 : (int8_t)-1;
}

__global__ void bn_softmax(const int32_t* __restrict__ acc,
                           float* __restrict__ out,
                           const float* __restrict__ beta,
                           const float* __restrict__ mean,
                           const float* __restrict__ var)
{
  const int r = threadIdx.x;
  if (r >= 32) return;
  float v[10];
  float mx = -3.4e38f;
  #pragma unroll
  for (int i = 0; i < 10; ++i) {
    const float y =
        ((float)acc[r * 64 + i] - mean[i]) * rsqrtf(var[i] + BN_EPS) + beta[i];
    v[i] = y;
    mx = fmaxf(mx, y);
  }
  float s = 0.f;
  #pragma unroll
  for (int i = 0; i < 10; ++i) { v[i] = __expf(v[i] - mx); s += v[i]; }
  const float inv = 1.f / s;
  #pragma unroll
  for (int i = 0; i < 10; ++i) out[r * 10 + i] = v[i] * inv;
}

extern "C" void kernel_launch(void* const* d_in, const int* in_sizes, int n_in,
                              void* d_out, int out_size, void* d_ws, size_t ws_size,
                              hipStream_t stream)
{
  (void)in_sizes; (void)n_in; (void)out_size; (void)ws_size;

  const float* x = (const float*)d_in[0];
  const float* cw[6]; for (int i = 0; i < 6; ++i) cw[i] = (const float*)d_in[1 + i];
  const float* fw[3]; for (int i = 0; i < 3; ++i) fw[i] = (const float*)d_in[7 + i];
  const float *bnB[9], *bnM[9], *bnV[9];
  for (int i = 0; i < 9; ++i) {
    bnB[i] = (const float*)d_in[10 + 3 * i];
    bnM[i] = (const float*)d_in[11 + 3 * i];
    bnV[i] = (const float*)d_in[12 + 3 * i];
  }

  char* ws = (char*)d_ws;
  const size_t MB = 1024 * 1024;
  const size_t KB = 1024;
  int8_t*  actA = (int8_t*)(ws + 0);
  int8_t*  actB = (int8_t*)(ws + 40 * MB);
  int16_t* acc  = (int16_t*)(ws + 56 * MB);
  int8_t*  wc2  = (int8_t*)(ws + 132 * MB +    0 * KB);
  int8_t*  wc3  = (int8_t*)(ws + 132 * MB +  144 * KB);
  int8_t*  wc4  = (int8_t*)(ws + 132 * MB +  432 * KB);
  int8_t*  wc5  = (int8_t*)(ws + 132 * MB + 1008 * KB);
  int8_t*  wc6  = (int8_t*)(ws + 132 * MB + 2160 * KB);
  int8_t*  wfc1 = (int8_t*)(ws + 56 * MB);
  int8_t*  wfc2 = (int8_t*)(ws + 138 * MB);
  int8_t*  wfc3 = (int8_t*)(ws + 139 * MB);
  int32_t* fca1 = (int32_t*)(ws + 140 * MB);
  int32_t* fca2 = (int32_t*)(ws + 140 * MB + 256 * KB);
  int32_t* fca3 = (int32_t*)(ws + 140 * MB + 512 * KB);

  auto TS = [&](const float* w, int8_t* o, int K, int N, int Npad) {
    const long total = (long)Npad * K;
    transpose_sign<<<(unsigned)((total + 255) / 256), 256, 0, stream>>>(w, o, K, N, total);
  };
  auto GEMM = [&](const int8_t* in, const int8_t* w, void* out, int M, int NP,
                  int H, int W, int Cin, int ks, int o32) {
    dim3 g((M + 63) / 64, NP / 64);
    bgemm_wmma<<<g, 256, 0, stream>>>(in, w, out, M, NP, H, W, Cin, ks, o32);
  };

  TS(cw[1], wc2, 9 * 128, 128, 128);
  TS(cw[2], wc3, 9 * 128, 256, 256);
  TS(cw[3], wc4, 9 * 256, 256, 256);
  TS(cw[4], wc5, 9 * 256, 512, 512);
  TS(cw[5], wc6, 9 * 512, 512, 512);

  { const int total = 32 * 96 * 96 * 128;
    conv1_bn_sign<<<(total + 255) / 256, 256, 0, stream>>>(
        x, cw[0], bnB[0], bnM[0], bnV[0], actA, total); }
  GEMM(actA, wc2, acc, 32 * 96 * 96, 128, 96, 96, 128, 3, 0);
  { const int total = 32 * 48 * 48 * 128;
    pool_bn_sign<<<(total + 255) / 256, 256, 0, stream>>>(
        acc, actB, bnB[1], bnM[1], bnV[1], 48, 48, 128, total); }

  GEMM(actB, wc3, acc, 32 * 48 * 48, 256, 48, 48, 128, 3, 0);
  { const int total = 32 * 48 * 48 * 256;
    bn_sign16<<<(total + 255) / 256, 256, 0, stream>>>(
        acc, actA, bnB[2], bnM[2], bnV[2], 256, total); }
  GEMM(actA, wc4, acc, 32 * 48 * 48, 256, 48, 48, 256, 3, 0);
  { const int total = 32 * 24 * 24 * 256;
    pool_bn_sign<<<(total + 255) / 256, 256, 0, stream>>>(
        acc, actB, bnB[3], bnM[3], bnV[3], 24, 24, 256, total); }

  GEMM(actB, wc5, acc, 32 * 24 * 24, 512, 24, 24, 256, 3, 0);
  { const int total = 32 * 24 * 24 * 512;
    bn_sign16<<<(total + 255) / 256, 256, 0, stream>>>(
        acc, actA, bnB[4], bnM[4], bnV[4], 512, total); }
  GEMM(actA, wc6, acc, 32 * 24 * 24, 512, 24, 24, 512, 3, 0);
  { const int total = 32 * 12 * 12 * 512;
    pool_bn_sign<<<(total + 255) / 256, 256, 0, stream>>>(
        acc, actB, bnB[5], bnM[5], bnV[5], 12, 12, 512, total); }

  TS(fw[0], wfc1, 12 * 12 * 512, 1024, 1024);
  TS(fw[1], wfc2, 1024, 1024, 1024);
  TS(fw[2], wfc3, 1024, 10, 64);

  GEMM(actB, wfc1, fca1, 32, 1024, 1, 1, 12 * 12 * 512, 1, 1);
  { const int total = 32 * 1024;
    bn_sign32<<<(total + 255) / 256, 256, 0, stream>>>(
        fca1, actA, bnB[6], bnM[6], bnV[6], 1024, total); }
  GEMM(actA, wfc2, fca2, 32, 1024, 1, 1, 1024, 1, 1);
  { const int total = 32 * 1024;
    bn_sign32<<<(total + 255) / 256, 256, 0, stream>>>(
        fca2, actB, bnB[7], bnM[7], bnV[7], 1024, total); }
  GEMM(actB, wfc3, fca3, 32, 64, 1, 1, 1024, 1, 1);

  bn_softmax<<<1, 32, 0, stream>>>(fca3, (float*)d_out, bnB[8], bnM[8], bnV[8]);
}
